// TwiRGCN_50190987821504
// MI455X (gfx1250) — matrix operand order repaired and checked by reading, not compiled, not run on hardware
//
#include <hip/hip_runtime.h>
#include <hip/hip_bf16.h>
#include <math.h>

#define NUM_ENTITIES 100000
#define NUM_TIMES    10000
#define D_DIM        512
#define DB_DIM       768
#define NUM_BASES    4
#define N_NODES      30000
#define N_EDGES      200000
#define BATCH        128
#define P_ENT_N      64
#define P_TIME_N     32
#define T_UNIQ_N     2000
#define V_TOTAL      (NUM_ENTITIES + NUM_TIMES)
#define N_CAT        (NUM_BASES * D_DIM + D_DIM)   // 2560
#define LDS_STRIDE   40                            // ushorts per LDS tile row (80B, 16B aligned)

// ---------------------------------------------------------------- bf16 helpers
typedef __bf16 bf16_t;
typedef bf16_t v16bf __attribute__((ext_vector_type(16)));
typedef float  v8f   __attribute__((ext_vector_type(8)));

union Frag16 { unsigned int u[8]; v16bf v; };

__device__ __forceinline__ unsigned short f32_to_bf16(float f) {
    unsigned int u = __float_as_uint(f);
    u += 0x7FFFu + ((u >> 16) & 1u);          // round-to-nearest-even
    return (unsigned short)(u >> 16);
}
__device__ __forceinline__ unsigned int pack_bf16(float a, float b) {
    return (unsigned int)f32_to_bf16(a) | ((unsigned int)f32_to_bf16(b) << 16);
}
__device__ __forceinline__ float bf16lo_to_f32(unsigned int u) {
    return __uint_as_float(u << 16);
}
__device__ __forceinline__ float bf16hi_to_f32(unsigned int u) {
    return __uint_as_float(u & 0xFFFF0000u);
}

// gfx1250 async global->LDS copy (16B per lane), tracked by ASYNCcnt
__device__ __forceinline__ void async_b128(unsigned lds_off, const unsigned short* g) {
    asm volatile("global_load_async_to_lds_b128 %0, %1, off"
                 :: "v"(lds_off), "v"(g) : "memory");
}
__device__ __forceinline__ void wait_async0() {
    asm volatile("s_wait_asynccnt 0x0" ::: "memory");
}

// ---------------------------------------------------------------- bert projections
__global__ __launch_bounds__(256)
void k_bert_proj(const float* __restrict__ ques,
                 const float* __restrict__ W_lin,  const float* __restrict__ b_lin,
                 const float* __restrict__ W_attn, const float* __restrict__ b_attn,
                 const float* __restrict__ W_prob, const float* __restrict__ b_prob,
                 float* __restrict__ q_lin, float* __restrict__ q_attn,
                 float* __restrict__ p_ent)
{
    __shared__ float sq[DB_DIM];
    __shared__ float red[256];
    const int b = blockIdx.x;
    const int t = threadIdx.x;
    sq[t]       = ques[b * DB_DIM + t];
    sq[t + 256] = ques[b * DB_DIM + t + 256];
    sq[t + 512] = ques[b * DB_DIM + t + 512];
    __syncthreads();

    float l0 = 0.f, l1 = 0.f, a0 = 0.f, a1 = 0.f;
    for (int k = 0; k < DB_DIM; ++k) {
        const float q = sq[k];
        l0 += q * W_lin [k * D_DIM + t];
        l1 += q * W_lin [k * D_DIM + t + 256];
        a0 += q * W_attn[k * D_DIM + t];
        a1 += q * W_attn[k * D_DIM + t + 256];
    }
    q_lin [b * D_DIM + t]       = l0 + b_lin[t];
    q_lin [b * D_DIM + t + 256] = l1 + b_lin[t + 256];
    q_attn[b * D_DIM + t]       = a0 + b_attn[t];
    q_attn[b * D_DIM + t + 256] = a1 + b_attn[t + 256];

    float pp = sq[t] * W_prob[t] + sq[t + 256] * W_prob[t + 256] + sq[t + 512] * W_prob[t + 512];
    red[t] = pp;
    __syncthreads();
    for (int s = 128; s > 0; s >>= 1) {
        if (t < s) red[t] += red[t + s];
        __syncthreads();
    }
    if (t == 0) p_ent[b] = 1.f / (1.f + __expf(-(red[0] + b_prob[0])));
}

// ---------------------------------------------------------------- weight pre-transpose+convert
// Wt[n][k] = bf16( [basis0|basis1|basis2|basis3|root][k][n] ), n in [0,2560), k in [0,512)
__global__ __launch_bounds__(256)
void k_cvt_w(const float* __restrict__ basis, const float* __restrict__ root,
             unsigned short* __restrict__ Wt)
{
    const int k = blockIdx.x;                 // 0..511
    for (int n = threadIdx.x; n < N_CAT; n += 256) {
        const float w = (n < NUM_BASES * D_DIM)
            ? basis[(size_t)(n >> 9) * (D_DIM * D_DIM) + (size_t)k * D_DIM + (n & 511)]
            : root[(size_t)k * D_DIM + (n - NUM_BASES * D_DIM)];
        Wt[(size_t)n * D_DIM + k] = f32_to_bf16(w);
    }
}

// ---------------------------------------------------------------- row gather/relu -> bf16
// dst[r] = bf16( maybe_relu( src[gather ? gather[r] : r] ) ), rows of 512
__global__ __launch_bounds__(256)
void k_cvt_rows(const float* __restrict__ src, const int* __restrict__ gather,
                unsigned short* __restrict__ dstb, int nrows, int do_relu)
{
    const int lane = threadIdx.x & 31;
    const int row  = blockIdx.x * 8 + (threadIdx.x >> 5);
    if (row >= nrows) return;
    const int srow = gather ? gather[row] : row;
    const float4* p = (const float4*)(src + (size_t)srow * D_DIM + lane * 16);
    unsigned int pk[8];
#pragma unroll
    for (int i = 0; i < 4; ++i) {
        float4 v = p[i];
        if (do_relu) {
            v.x = fmaxf(v.x, 0.f); v.y = fmaxf(v.y, 0.f);
            v.z = fmaxf(v.z, 0.f); v.w = fmaxf(v.w, 0.f);
        }
        pk[2 * i]     = pack_bf16(v.x, v.y);
        pk[2 * i + 1] = pack_bf16(v.z, v.w);
    }
    uint4* o = (uint4*)(dstb + (size_t)row * D_DIM + lane * 16);
    uint4 u0; u0.x = pk[0]; u0.y = pk[1]; u0.z = pk[2]; u0.w = pk[3];
    uint4 u1; u1.x = pk[4]; u1.y = pk[5]; u1.z = pk[6]; u1.w = pk[7];
    o[0] = u0; o[1] = u1;
}

// ---------------------------------------------------------------- emb -> bf16 + row norms (fused)
__global__ __launch_bounds__(256)
void k_cvt_emb_norms(const float* __restrict__ emb, unsigned short* __restrict__ Eb,
                     float* __restrict__ norms)
{
    const int lane = threadIdx.x & 31;
    const int row  = blockIdx.x * 8 + (threadIdx.x >> 5);
    if (row >= V_TOTAL) return;
    const float4* p = (const float4*)(emb + (size_t)row * D_DIM + lane * 16);
    float s = 0.f;
    unsigned int pk[8];
#pragma unroll
    for (int i = 0; i < 4; ++i) {
        const float4 v = p[i];
        s += v.x * v.x + v.y * v.y + v.z * v.z + v.w * v.w;
        pk[2 * i]     = pack_bf16(v.x, v.y);
        pk[2 * i + 1] = pack_bf16(v.z, v.w);
    }
    uint4* o = (uint4*)(Eb + (size_t)row * D_DIM + lane * 16);
    uint4 u0; u0.x = pk[0]; u0.y = pk[1]; u0.z = pk[2]; u0.w = pk[3];
    uint4 u1; u1.x = pk[4]; u1.y = pk[5]; u1.z = pk[6]; u1.w = pk[7];
    o[0] = u0; o[1] = u1;
#pragma unroll
    for (int off = 16; off > 0; off >>= 1) s += __shfl_xor(s, off, 32);
    if (lane == 0) norms[row] = sqrtf(s);
}

// ---------------------------------------------------------------- fused basis+root GEMM (WMMA bf16)
// C[M x 2560] = Ab[M x 512] @ Wt^T ; cols [0,2048)->xb bf16, [2048,2560)->agg=+bias f32
// Block tile 128x64, 8 waves as 4(M) x 2(N); tiles staged with async global->LDS b128.
__global__ __launch_bounds__(256)
void k_gemm_xb(const unsigned short* __restrict__ Ab, const unsigned short* __restrict__ Wt,
               const float* __restrict__ bias,
               unsigned short* __restrict__ xb, float* __restrict__ agg)
{
    __shared__ unsigned short As[128 * LDS_STRIDE];
    __shared__ unsigned short Bt[64 * LDS_STRIDE];

    const int tid  = threadIdx.x;
    const int lane = tid & 31;
    const int wave = tid >> 5;
    const int wm   = wave >> 1;
    const int wn   = wave & 1;
    const int mBase = blockIdx.y * 128;
    const int nBase = blockIdx.x * 64;

    const unsigned asB = (unsigned)(uintptr_t)(void*)As;
    const unsigned btB = (unsigned)(uintptr_t)(void*)Bt;

    v8f acc[2][2] = {};

    for (int k0 = 0; k0 < D_DIM; k0 += 32) {
        // A tile: 128 rows x 64B -> 512 x b128 chunks, 2 per thread
#pragma unroll
        for (int it = 0; it < 2; ++it) {
            const int q = tid + 256 * it;
            const int r = q >> 2;
            const int c = q & 3;
            int gm = mBase + r;
            gm = (gm < N_NODES) ? gm : 0;
            async_b128(asB + (unsigned)(r * (LDS_STRIDE * 2) + c * 16),
                       Ab + (size_t)gm * D_DIM + k0 + c * 8);
        }
        // B tile: 64 rows x 64B -> 256 x b128 chunks, 1 per thread
        {
            const int r = tid >> 2;
            const int c = tid & 3;
            async_b128(btB + (unsigned)(r * (LDS_STRIDE * 2) + c * 16),
                       Wt + (size_t)(nBase + r) * D_DIM + k0 + c * 8);
        }
        wait_async0();
        __syncthreads();

        const int hi16 = lane >> 4;
        const int mrow = wm * 32 + (lane & 15);
        const int ncol = wn * 32 + (lane & 15);
        Frag16 fa[2], fb[2];
#pragma unroll
        for (int i = 0; i < 2; ++i)
#pragma unroll
            for (int v = 0; v < 8; ++v) {
                const int kb = (v < 4) ? (2 * v + 8 * hi16) : (2 * v + 8 + 8 * hi16);
                fa[i].u[v] = *(const unsigned int*)(As + (mrow + i * 16) * LDS_STRIDE + kb);
            }
#pragma unroll
        for (int j = 0; j < 2; ++j)
#pragma unroll
            for (int v = 0; v < 8; ++v) {
                const int kb = (v < 4) ? (2 * v + 8 * hi16) : (2 * v + 8 + 8 * hi16);
                fb[j].u[v] = *(const unsigned int*)(Bt + (ncol + j * 16) * LDS_STRIDE + kb);
            }
#pragma unroll
        for (int i = 0; i < 2; ++i)
#pragma unroll
            for (int j = 0; j < 2; ++j)
                acc[i][j] = __builtin_amdgcn_wmma_f32_16x16x32_bf16(
                    false, fa[i].v, false, fb[j].v, (short)0, acc[i][j], false, false);
        __syncthreads();
    }

    const int hi16 = lane >> 4;
    const int nLoc = lane & 15;
#pragma unroll
    for (int i = 0; i < 2; ++i)
#pragma unroll
        for (int j = 0; j < 2; ++j) {
            const int gn = nBase + wn * 32 + j * 16 + nLoc;
#pragma unroll
            for (int v = 0; v < 8; ++v) {
                const int gm = mBase + wm * 32 + i * 16 + v + 8 * hi16;
                if (gm < N_NODES) {
                    const float val = acc[i][j][v];
                    if (gn < NUM_BASES * D_DIM)
                        xb[(size_t)gm * (NUM_BASES * D_DIM) + gn] = f32_to_bf16(val);
                    else
                        agg[(size_t)gm * D_DIM + (gn - NUM_BASES * D_DIM)] =
                            val + bias[gn - NUM_BASES * D_DIM];
                }
            }
        }
}

// ---------------------------------------------------------------- edge messages (wave per edge)
__global__ __launch_bounds__(256)
void k_edge(const unsigned short* __restrict__ xb, float* __restrict__ agg,
            const float* __restrict__ att,
            const int* __restrict__ edge_index, const int* __restrict__ edge_type,
            const int* __restrict__ st, const int* __restrict__ et,
            const int* __restrict__ ebi, const float* __restrict__ enorm,
            const float* __restrict__ q_attn, const float* __restrict__ emb)
{
    const int lane = threadIdx.x & 31;
    const int e = blockIdx.x * 8 + (threadIdx.x >> 5);
    if (e >= N_EDGES) return;

    const int src  = edge_index[e];
    const int dst  = edge_index[N_EDGES + e];
    const int type = edge_type[e];

    const float* qrow = q_attn + (size_t)ebi[e] * D_DIM;
    const float* srow = emb + (size_t)(NUM_ENTITIES + st[e]) * D_DIM;
    const float* trow = emb + (size_t)(NUM_ENTITIES + et[e]) * D_DIM;

    float dot = 0.f;
    const float4* q4 = (const float4*)(qrow + lane * 16);
    const float4* s4 = (const float4*)(srow + lane * 16);
    const float4* t4 = (const float4*)(trow + lane * 16);
#pragma unroll
    for (int i = 0; i < 4; ++i) {
        const float4 a = q4[i], b1 = s4[i], b2 = t4[i];
        dot += a.x * (b1.x + b2.x) + a.y * (b1.y + b2.y) +
               a.z * (b1.z + b2.z) + a.w * (b1.w + b2.w);
    }
    dot *= 0.5f;
#pragma unroll
    for (int off = 16; off > 0; off >>= 1) dot += __shfl_xor(dot, off, 32);
    const float w = (1.f / (1.f + __expf(-dot))) * enorm[e];

    const unsigned short* row = xb + (size_t)src * (NUM_BASES * D_DIM) + lane * 16;
    float* outp = agg + (size_t)dst * D_DIM + lane * 16;

    float msg[16];
#pragma unroll
    for (int i = 0; i < 16; ++i) msg[i] = 0.f;
#pragma unroll
    for (int bb = 0; bb < NUM_BASES; ++bb) {
        const float c = att[type * NUM_BASES + bb] * w;
        const uint4* p = (const uint4*)(row + bb * D_DIM);
        const uint4 lo = p[0], hi = p[1];
        const unsigned int u[8] = {lo.x, lo.y, lo.z, lo.w, hi.x, hi.y, hi.z, hi.w};
#pragma unroll
        for (int qq = 0; qq < 8; ++qq) {
            msg[2 * qq]     += c * bf16lo_to_f32(u[qq]);
            msg[2 * qq + 1] += c * bf16hi_to_f32(u[qq]);
        }
    }
#pragma unroll
    for (int i = 0; i < 16; ++i) unsafeAtomicAdd(outp + i, msg[i]);
}

// ---------------------------------------------------------------- pooling + pred + normalize -> bf16
__global__ __launch_bounds__(256)
void k_pool_pred(const float* __restrict__ h, const float* __restrict__ emb,
                 const int* __restrict__ uniq_times,
                 const int* __restrict__ padded_ent, const int* __restrict__ padded_time,
                 const int* __restrict__ ent_counts, const int* __restrict__ time_counts,
                 const float* __restrict__ q_lin, const float* __restrict__ p_ent,
                 unsigned short* __restrict__ prednb)
{
    __shared__ float red[256];
    const int b = blockIdx.x;
    const int t = threadIdx.x;

    float pe0 = 0.f, pe1 = 0.f;
    for (int j = 0; j < P_ENT_N; ++j) {
        const int idx = padded_ent[b * P_ENT_N + j];
        if (idx < N_NODES) {
            pe0 += h[(size_t)idx * D_DIM + t];
            pe1 += h[(size_t)idx * D_DIM + t + 256];
        }
    }
    float pt0 = 0.f, pt1 = 0.f;
    for (int j = 0; j < P_TIME_N; ++j) {
        const int idx = padded_time[b * P_TIME_N + j];
        if (idx < T_UNIQ_N) {
            const size_t r = (size_t)(NUM_ENTITIES + uniq_times[idx]) * D_DIM;
            pt0 += emb[r + t];
            pt1 += emb[r + t + 256];
        }
    }
    const float ec = (float)ent_counts[b];
    const float tc = (float)time_counts[b];
    pe0 /= ec; pe1 /= ec; pt0 /= tc; pt1 /= tc;

    const float p = p_ent[b];
    const float v0 = (q_lin[b * D_DIM + t]       + pe0 * p + pt0 * (1.f - p)) * (1.f / 3.f);
    const float v1 = (q_lin[b * D_DIM + t + 256] + pe1 * p + pt1 * (1.f - p)) * (1.f / 3.f);

    red[t] = v0 * v0 + v1 * v1;
    __syncthreads();
    for (int s = 128; s > 0; s >>= 1) {
        if (t < s) red[t] += red[t + s];
        __syncthreads();
    }
    const float inv = 1.f / fmaxf(sqrtf(red[0]), 1e-8f);
    prednb[b * D_DIM + t]       = f32_to_bf16(v0 * inv);
    prednb[b * D_DIM + t + 256] = f32_to_bf16(v1 * inv);
}

// ---------------------------------------------------------------- score GEMM (WMMA bf16, async tiles)
// out[128 x V] = 30 * (pred_n @ emb^T) / max(||emb||,1e-8)
__global__ __launch_bounds__(256)
void k_gemm_score(const unsigned short* __restrict__ prednb, const unsigned short* __restrict__ Eb,
                  const float* __restrict__ norms, float* __restrict__ outp)
{
    __shared__ unsigned short As[128 * LDS_STRIDE];
    __shared__ unsigned short Bt[64 * LDS_STRIDE];

    const int tid  = threadIdx.x;
    const int lane = tid & 31;
    const int wave = tid >> 5;
    const int wm   = wave >> 1;
    const int wn   = wave & 1;
    const int nBase = blockIdx.x * 64;

    const unsigned asB = (unsigned)(uintptr_t)(void*)As;
    const unsigned btB = (unsigned)(uintptr_t)(void*)Bt;

    v8f acc[2][2] = {};

    for (int k0 = 0; k0 < D_DIM; k0 += 32) {
#pragma unroll
        for (int it = 0; it < 2; ++it) {
            const int q = tid + 256 * it;
            const int r = q >> 2;      // 0..127 (M=128 exact)
            const int c = q & 3;
            async_b128(asB + (unsigned)(r * (LDS_STRIDE * 2) + c * 16),
                       prednb + (size_t)r * D_DIM + k0 + c * 8);
        }
        {
            const int r = tid >> 2;
            const int c = tid & 3;
            int gn = nBase + r;
            gn = (gn < V_TOTAL) ? gn : 0;
            async_b128(btB + (unsigned)(r * (LDS_STRIDE * 2) + c * 16),
                       Eb + (size_t)gn * D_DIM + k0 + c * 8);
        }
        wait_async0();
        __syncthreads();

        const int hi16 = lane >> 4;
        const int mrow = wm * 32 + (lane & 15);
        const int ncol = wn * 32 + (lane & 15);
        Frag16 fa[2], fb[2];
#pragma unroll
        for (int i = 0; i < 2; ++i)
#pragma unroll
            for (int v = 0; v < 8; ++v) {
                const int kb = (v < 4) ? (2 * v + 8 * hi16) : (2 * v + 8 + 8 * hi16);
                fa[i].u[v] = *(const unsigned int*)(As + (mrow + i * 16) * LDS_STRIDE + kb);
            }
#pragma unroll
        for (int j = 0; j < 2; ++j)
#pragma unroll
            for (int v = 0; v < 8; ++v) {
                const int kb = (v < 4) ? (2 * v + 8 * hi16) : (2 * v + 8 + 8 * hi16);
                fb[j].u[v] = *(const unsigned int*)(Bt + (ncol + j * 16) * LDS_STRIDE + kb);
            }
#pragma unroll
        for (int i = 0; i < 2; ++i)
#pragma unroll
            for (int j = 0; j < 2; ++j)
                acc[i][j] = __builtin_amdgcn_wmma_f32_16x16x32_bf16(
                    false, fa[i].v, false, fb[j].v, (short)0, acc[i][j], false, false);
        __syncthreads();
    }

    const int hi16 = lane >> 4;
    const int nLoc = lane & 15;
#pragma unroll
    for (int i = 0; i < 2; ++i)
#pragma unroll
        for (int j = 0; j < 2; ++j) {
            const int gn = nBase + wn * 32 + j * 16 + nLoc;
            if (gn < V_TOTAL) {
                const float sc = 30.f / fmaxf(norms[gn], 1e-8f);
#pragma unroll
                for (int v = 0; v < 8; ++v) {
                    const int gm = wm * 32 + i * 16 + v + 8 * hi16;   // < 128 always
                    outp[(size_t)gm * V_TOTAL + gn] = acc[i][j][v] * sc;
                }
            }
        }
}

// ---------------------------------------------------------------- host launcher
extern "C" void kernel_launch(void* const* d_in, const int* in_sizes, int n_in,
                              void* d_out, int out_size, void* d_ws, size_t ws_size,
                              hipStream_t stream)
{
    (void)in_sizes; (void)n_in; (void)out_size; (void)ws_size;

    const float* emb        = (const float*)d_in[0];
    const float* basis1     = (const float*)d_in[1];
    const float* att1       = (const float*)d_in[2];
    const float* root1      = (const float*)d_in[3];
    const float* bias1      = (const float*)d_in[4];
    const float* basis2     = (const float*)d_in[5];
    const float* att2       = (const float*)d_in[6];
    const float* root2      = (const float*)d_in[7];
    const float* bias2      = (const float*)d_in[8];
    const float* W_lin      = (const float*)d_in[9];
    const float* b_lin      = (const float*)d_in[10];
    const float* W_attn     = (const float*)d_in[11];
    const float* b_attn     = (const float*)d_in[12];
    const float* W_prob     = (const float*)d_in[13];
    const float* b_prob     = (const float*)d_in[14];
    const float* ques       = (const float*)d_in[15];
    const int*   x_idx      = (const int*)d_in[16];
    const int*   edge_index = (const int*)d_in[17];
    const int*   edge_type  = (const int*)d_in[18];
    const int*   start_time = (const int*)d_in[19];
    const int*   end_time   = (const int*)d_in[20];
    const int*   edge_batch = (const int*)d_in[21];
    const float* edge_norm  = (const float*)d_in[22];
    const int*   uniq_times = (const int*)d_in[23];
    const int*   padded_ent = (const int*)d_in[24];
    const int*   padded_time= (const int*)d_in[25];
    const int*   ent_counts = (const int*)d_in[26];
    const int*   time_counts= (const int*)d_in[27];
    float* out = (float*)d_out;

    char* ws = (char*)d_ws;
    size_t off = 0;
    auto take = [&](size_t bytes) -> void* {
        void* p = ws + off;
        off = (off + bytes + 255) & ~(size_t)255;
        return p;
    };
    float*          q_lin   = (float*)take((size_t)BATCH * D_DIM * 4);
    float*          q_attn  = (float*)take((size_t)BATCH * D_DIM * 4);
    float*          p_ent   = (float*)take((size_t)BATCH * 4);
    unsigned short* Wt1     = (unsigned short*)take((size_t)N_CAT * D_DIM * 2);
    unsigned short* Wt2     = (unsigned short*)take((size_t)N_CAT * D_DIM * 2);
    unsigned short* Ab      = (unsigned short*)take((size_t)N_NODES * D_DIM * 2);
    unsigned short* xb      = (unsigned short*)take((size_t)N_NODES * NUM_BASES * D_DIM * 2);
    float*          agg     = (float*)take((size_t)N_NODES * D_DIM * 4);
    unsigned short* Eb      = (unsigned short*)take((size_t)V_TOTAL * D_DIM * 2);
    float*          norms   = (float*)take((size_t)V_TOTAL * 4);
    unsigned short* prednb  = (unsigned short*)take((size_t)BATCH * D_DIM * 2);
    (void)take(65536);   // slack

    const dim3 gemmGrid(N_CAT / 64, (N_NODES + 127) / 128);   // 40 x 235

    k_bert_proj<<<BATCH, 256, 0, stream>>>(ques, W_lin, b_lin, W_attn, b_attn,
                                           W_prob, b_prob, q_lin, q_attn, p_ent);
    k_cvt_w<<<D_DIM, 256, 0, stream>>>(basis1, root1, Wt1);
    k_cvt_w<<<D_DIM, 256, 0, stream>>>(basis2, root2, Wt2);
    k_cvt_emb_norms<<<(V_TOTAL + 7) / 8, 256, 0, stream>>>(emb, Eb, norms);

    // layer 1
    k_cvt_rows<<<(N_NODES + 7) / 8, 256, 0, stream>>>(emb, x_idx, Ab, N_NODES, 0);
    k_gemm_xb<<<gemmGrid, 256, 0, stream>>>(Ab, Wt1, bias1, xb, agg);
    k_edge<<<N_EDGES / 8, 256, 0, stream>>>(xb, agg, att1, edge_index, edge_type,
                                            start_time, end_time, edge_batch,
                                            edge_norm, q_attn, emb);
    // relu -> bf16 A for layer 2
    k_cvt_rows<<<(N_NODES + 7) / 8, 256, 0, stream>>>(agg, nullptr, Ab, N_NODES, 1);

    // layer 2
    k_gemm_xb<<<gemmGrid, 256, 0, stream>>>(Ab, Wt2, bias2, xb, agg);
    k_edge<<<N_EDGES / 8, 256, 0, stream>>>(xb, agg, att2, edge_index, edge_type,
                                            start_time, end_time, edge_batch,
                                            edge_norm, q_attn, emb);

    // readout
    k_pool_pred<<<BATCH, 256, 0, stream>>>(agg, emb, uniq_times, padded_ent, padded_time,
                                           ent_counts, time_counts, q_lin, p_ent, prednb);
    k_gemm_score<<<(V_TOTAL + 63) / 64, 256, 0, stream>>>(prednb, Eb, norms, out);
}
